// Self_dimMultiHead_Attention1_35536559407617
// MI455X (gfx1250) — matrix operand order repaired and checked by reading, not compiled, not run on hardware
//
#include <hip/hip_runtime.h>

#define Bb  4
#define Ll  2048
#define Ee  1024
#define Hh  8
#define Dd  128
#define OO  1024

typedef __attribute__((ext_vector_type(16))) __bf16 v16bf;
typedef __attribute__((ext_vector_type(8)))  __bf16 v8bf;
typedef __attribute__((ext_vector_type(2)))  __bf16 v2bf;
typedef __attribute__((ext_vector_type(8)))  float  v8f;

static __device__ inline v8f vzero8f() {
  v8f z;
#pragma unroll
  for (int i = 0; i < 8; ++i) z[i] = 0.0f;
  return z;
}

// f32 -> bf16: native convert (backend selects v_cvt / packed form)
static __device__ inline __bf16 f2bf(float f) { return (__bf16)f; }

static __device__ inline v2bf f2bf2(float a, float b) {
#if __has_builtin(__builtin_amdgcn_cvt_pk_bf16_f32)
  return __builtin_amdgcn_cvt_pk_bf16_f32(a, b);
#else
  v2bf r; r[0] = (__bf16)a; r[1] = (__bf16)b;
  return r;
#endif
}

static __device__ inline float fast_rcp(float x) {
  return __builtin_amdgcn_rcpf(x);
}

// assemble a 16-half WMMA fragment from two 16-byte chunks
static __device__ inline v16bf load_frag16(const __bf16* p0, const __bf16* p1) {
  v16bf r;
  *(v8bf*)&r       = *(const v8bf*)p0;
  *((v8bf*)&r + 1) = *(const v8bf*)p1;
  return r;
}

#define WMMA_BF16 __builtin_amdgcn_wmma_f32_16x16x32_bf16

// ---------------------------------------------------------------------------
// Tensor Data Mover: 2D tile load (global -> LDS), ISA 08_async_tensor §8.
// This toolchain (clang-23/therock) uses the 6-arg builtin form.
// ---------------------------------------------------------------------------
#if __has_builtin(__builtin_amdgcn_tensor_load_to_lds)
#define HAVE_TDM 1
typedef unsigned v4u __attribute__((ext_vector_type(4)));
typedef int      v8i __attribute__((ext_vector_type(8)));
typedef int      v4i __attribute__((ext_vector_type(4)));

// 2-byte elements; tile_x elems per row, tile_y rows, row stride in elems.
static __device__ inline void tdm_load_2d(unsigned lds_addr, const void* gaddr,
                                          unsigned tile_x, unsigned tile_y,
                                          unsigned stride_elems) {
  unsigned long ga = (unsigned long)gaddr;
  v4u g0;
  g0[0] = 1u;                                   // count=1, user descriptor
  g0[1] = lds_addr;                             // D#.lds_addr (bytes)
  g0[2] = (unsigned)(ga & 0xFFFFFFFFu);         // global_addr[31:0]
  g0[3] = (unsigned)((ga >> 32) & 0x01FFFFFFu)  // global_addr[56:32]
          | (2u << 30);                         // type = 2 ("image")
  v8i g1;
  g1[0] = 0x00010000;                                        // data_size=1 (2B)
  g1[1] = (int)((tile_x & 0xFFFFu) << 16);                   // tensor_dim0[15:0]
  g1[2] = (int)(((tile_x >> 16) & 0xFFFFu)                   // tensor_dim0[31:16]
          | ((tile_y & 0xFFFFu) << 16));                     // tensor_dim1[15:0]
  g1[3] = (int)(((tile_y >> 16) & 0xFFFFu)                   // tensor_dim1[31:16]
          | ((tile_x & 0xFFFFu) << 16));                     // tile_dim0
  g1[4] = (int)(tile_y & 0xFFFFu);                           // tile_dim1, tile_dim2=0
  g1[5] = (int)stride_elems;                                 // tensor_dim0_stride[31:0]
  g1[6] = 0;                                                 // stride0 hi, stride1 lo
  g1[7] = 0;
  v4i z4 = {0, 0, 0, 0};                                     // groups 2/3 unused (2D)
  v8i z8 = {0, 0, 0, 0, 0, 0, 0, 0};
  __builtin_amdgcn_tensor_load_to_lds(g0, g1, z4, z4, z8, 0);
}

static __device__ inline unsigned lds_offset(const void* p) {
  return (unsigned)(unsigned long)p;   // LDS aperture: addr[31:0] is the LDS byte offset
}
#else
#define HAVE_TDM 0
#endif

// ---------------------------------------------------------------------------
// Kernel 0: convert weights to bf16. w_kx/w_qx transposed [H,E,D]->[H,D,E].
// ---------------------------------------------------------------------------
__global__ void cvt_weights(const float* __restrict__ wk, const float* __restrict__ wq,
                            const float* __restrict__ pw,
                            __bf16* __restrict__ wkT, __bf16* __restrict__ wqT,
                            __bf16* __restrict__ pwb) {
  long i = (long)blockIdx.x * blockDim.x + threadIdx.x;
  const long HED = (long)Hh * Ee * Dd;
  const long OHD = (long)OO * Hh * Dd;
  if (i < HED) {
    int h = (int)(i / (Ee * Dd));
    int r = (int)(i % (Ee * Dd));
    int e = r / Dd, d = r % Dd;
    wkT[(long)h * Dd * Ee + (long)d * Ee + e] = f2bf(wk[i]);
  } else if (i < 2 * HED) {
    long j = i - HED;
    int h = (int)(j / (Ee * Dd));
    int r = (int)(j % (Ee * Dd));
    int e = r / Dd, d = r % Dd;
    wqT[(long)h * Dd * Ee + (long)d * Ee + e] = f2bf(wq[j]);
  } else if (i < 2 * HED + OHD) {
    long j = i - 2 * HED;
    pwb[j] = f2bf(pw[j]);
  }
}

// ---------------------------------------------------------------------------
// Kernel 1: y = sigmoid(x @ w[h]) per (b,h). Wave: 16 L-rows x 128 D.
// ---------------------------------------------------------------------------
__global__ void __launch_bounds__(256)
proj_act(const float* __restrict__ x,    // [B,L,E] f32
         const __bf16* __restrict__ wT,  // [H,D,E] bf16
         __bf16* __restrict__ y,         // [B*H, L, D]
         __bf16* __restrict__ yT,        // [B*H, D, L]
         int writeT) {
  int bh   = blockIdx.x >> 4;
  int tile = blockIdx.x & 15;
  int b = bh >> 3, h = bh & 7;
  int wave = threadIdx.x >> 5;
  int ln = threadIdx.x & 31;
  int lh = ln & 15;
  int hi = ln >> 4;

  const float*  xb = x  + (long)b * Ll * Ee;
  const __bf16* wh = wT + (long)h * Dd * Ee;
  int row0 = tile * 128 + wave * 16;

  v8f acc[8];
#pragma unroll
  for (int nb = 0; nb < 8; ++nb) acc[nb] = vzero8f();

  for (int e0 = 0; e0 < Ee; e0 += 32) {
    const float* ap = xb + (long)(row0 + lh) * Ee + e0 + hi * 8;
    v16bf a;
#pragma unroll
    for (int i = 0; i < 4; ++i) *((v2bf*)&a + i)     = f2bf2(ap[2*i],    ap[2*i+1]);
#pragma unroll
    for (int i = 0; i < 4; ++i) *((v2bf*)&a + 4 + i) = f2bf2(ap[16+2*i], ap[16+2*i+1]);

    v16bf bfr[8];
#pragma unroll
    for (int nb = 0; nb < 8; ++nb) {
      const __bf16* bp = wh + (long)(nb * 16 + lh) * Ee + e0 + hi * 8;
      bfr[nb] = load_frag16(bp, bp + 16);
    }
    acc[0] = WMMA_BF16(false, a, false, bfr[0], (short)0, acc[0], false, false);
#pragma unroll
    for (int nb = 1; nb < 8; ++nb)
      acc[nb] = WMMA_BF16(false, a, false, bfr[nb], (short)0, acc[nb], true, false);
  }

  __bf16* yb  = y  + (long)bh * Ll * Dd;
  __bf16* ytb = yT + (long)bh * Dd * Ll;
#pragma unroll
  for (int nb = 0; nb < 8; ++nb) {
#pragma unroll
    for (int r = 0; r < 8; ++r) {
      int row = row0 + r + hi * 8;
      int col = nb * 16 + lh;
      float v = fast_rcp(1.0f + __expf(-acc[nb][r]));
      __bf16 bv = f2bf(v);
      yb[(long)row * Dd + col] = bv;
      if (writeT) ytb[(long)col * Ll + row] = bv;
    }
  }
}

// ---------------------------------------------------------------------------
// Kernel 2: flash attention per (b,h). Block = 8 waves x 16 q-rows = 128 rows.
// K / V^T tiles staged once per block via TENSOR_LOAD_TO_LDS when available.
// ---------------------------------------------------------------------------
__global__ void __launch_bounds__(256)
attn_kernel(const __bf16* __restrict__ qx,   // [B*H, L, D]
            const __bf16* __restrict__ kx,   // [B*H, L, D]
            const __bf16* __restrict__ kxT,  // [B*H, D, L]
            __bf16* __restrict__ obuf)       // [B, L, H*D]
{
  __shared__ __bf16 Psh[8 * 16 * 64];        // 16 KB: per-wave P staging
#if HAVE_TDM
  __shared__ __bf16 Kt[64 * 128];            // 16 KB: K tile  [key][d]
  __shared__ __bf16 Vt[128 * 64];            // 16 KB: V^T tile [d][key]
#endif
  int bh    = blockIdx.x >> 4;
  int qtile = blockIdx.x & 15;
  int b = bh >> 3, h = bh & 7;
  int wave = threadIdx.x >> 5;
  int ln = threadIdx.x & 31;
  int lh = ln & 15, hi = ln >> 4;

  const __bf16* qb  = qx  + (long)bh * Ll * Dd;
  const __bf16* kb  = kx  + (long)bh * Ll * Dd;
  const __bf16* ktb = kxT + (long)bh * Dd * Ll;
  __bf16* Pw = Psh + wave * (16 * 64);

  int qrow0 = qtile * 128 + wave * 16;
  const float scale = 0.08838834764831845f; // 1/sqrt(128)

  v16bf qf[4];
#pragma unroll
  for (int ks = 0; ks < 4; ++ks) {
    const __bf16* p = qb + (long)(qrow0 + lh) * Dd + ks * 32 + hi * 8;
    qf[ks] = load_frag16(p, p + 16);
  }

  v8f acc[8];
#pragma unroll
  for (int nb = 0; nb < 8; ++nb) acc[nb] = vzero8f();
  float m[8], l[8];
#pragma unroll
  for (int r = 0; r < 8; ++r) { m[r] = -3.0e38f; l[r] = 0.0f; }

  for (int kt = 0; kt < Ll / 64; ++kt) {
    int key0 = kt * 64;
    if (kt + 1 < Ll / 64)
      __builtin_prefetch(kb + (long)(key0 + 64 + ln) * Dd, 0, 0);

#if HAVE_TDM
    // stage K (64x128) and V^T (128x64) tiles into LDS once per block
    __syncthreads();                    // previous iteration done with Kt/Vt
    if (wave == 0) {
      tdm_load_2d(lds_offset(Kt), kb + (long)key0 * Dd, Dd, 64, Dd);
      tdm_load_2d(lds_offset(Vt), ktb + key0,          64, Dd, Ll);
      __builtin_amdgcn_s_wait_tensorcnt(0);
    }
    __syncthreads();                    // tiles visible to all waves
#endif

    // ---- scores S = Q * K^T (16 x 64), f32 accum
    v8f s[4];
#pragma unroll
    for (int nb = 0; nb < 4; ++nb) s[nb] = vzero8f();
#pragma unroll
    for (int ks = 0; ks < 4; ++ks) {
      v16bf kf[4];
#pragma unroll
      for (int nb = 0; nb < 4; ++nb) {
#if HAVE_TDM
        const __bf16* bp = Kt + (nb * 16 + lh) * Dd + ks * 32 + hi * 8;
#else
        const __bf16* bp = kb + (long)(key0 + nb * 16 + lh) * Dd + ks * 32 + hi * 8;
#endif
        kf[nb] = load_frag16(bp, bp + 16);
      }
      s[0] = WMMA_BF16(false, qf[ks], false, kf[0], (short)0, s[0], false, false);
#pragma unroll
      for (int nb = 1; nb < 4; ++nb)
        s[nb] = WMMA_BF16(false, qf[ks], false, kf[nb], (short)0, s[nb], true, false);
    }

    // ---- online softmax (per row = r + hi*8)
#pragma unroll
    for (int r = 0; r < 8; ++r) {
      float mx = -3.0e38f;
#pragma unroll
      for (int nb = 0; nb < 4; ++nb) mx = fmaxf(mx, s[nb][r]);
      mx *= scale;
#pragma unroll
      for (int off = 1; off < 16; off <<= 1) mx = fmaxf(mx, __shfl_xor(mx, off, 32));
      float mnew = fmaxf(m[r], mx);
      float corr = __expf(m[r] - mnew);
      float rsum = 0.0f;
      int prow = r + hi * 8;
#pragma unroll
      for (int nb = 0; nb < 4; ++nb) {
        float p = __expf(s[nb][r] * scale - mnew);
        rsum += p;
        Pw[prow * 64 + nb * 16 + lh] = f2bf(p);
      }
#pragma unroll
      for (int off = 1; off < 16; off <<= 1) rsum += __shfl_xor(rsum, off, 32);
      l[r] = l[r] * corr + rsum;
      m[r] = mnew;
#pragma unroll
      for (int nb = 0; nb < 8; ++nb) acc[nb][r] *= corr;
    }

    // ---- O += P * V  (P re-read from LDS in A-layout)
#pragma unroll
    for (int kc = 0; kc < 2; ++kc) {
      const __bf16* pp = Pw + lh * 64 + kc * 32 + hi * 8;
      v16bf pf = load_frag16(pp, pp + 16);
      v16bf vf[8];
#pragma unroll
      for (int nb = 0; nb < 8; ++nb) {
#if HAVE_TDM
        const __bf16* bp = Vt + (nb * 16 + lh) * 64 + kc * 32 + hi * 8;
#else
        const __bf16* bp = ktb + (long)(nb * 16 + lh) * Ll + key0 + kc * 32 + hi * 8;
#endif
        vf[nb] = load_frag16(bp, bp + 16);
      }
      acc[0] = WMMA_BF16(false, pf, false, vf[0], (short)0, acc[0], false, false);
#pragma unroll
      for (int nb = 1; nb < 8; ++nb)
        acc[nb] = WMMA_BF16(false, pf, false, vf[nb], (short)0, acc[nb], true, false);
    }
  }

  // ---- normalize and write concat-head layout [B, L, H*D] in bf16
  __bf16* ob = obuf + (long)b * Ll * (Hh * Dd) + h * Dd;
#pragma unroll
  for (int r = 0; r < 8; ++r) {
    float inv = fast_rcp(l[r]);
    int row = qrow0 + r + hi * 8;
#pragma unroll
    for (int nb = 0; nb < 8; ++nb) {
      ob[(long)row * (Hh * Dd) + nb * 16 + lh] = f2bf(acc[nb][r] * inv);
    }
  }
}

// ---------------------------------------------------------------------------
// Kernel 3: out = obuf @ proj_w^T + proj_b.  [8192 x 1024] x [1024 x 1024].
// ---------------------------------------------------------------------------
__global__ void __launch_bounds__(256)
proj_kernel(const __bf16* __restrict__ a,    // [B*L, H*D] bf16
            const __bf16* __restrict__ wb,   // [OUT, H*D] bf16
            const float*  __restrict__ bias, // [OUT]
            float* __restrict__ out)         // [B*L, OUT] f32
{
  int mt = blockIdx.x >> 3;
  int nt = blockIdx.x & 7;
  int wave = threadIdx.x >> 5;
  int ln = threadIdx.x & 31;
  int lh = ln & 15, hi = ln >> 4;
  int row0 = mt * 128 + wave * 16;
  int col0 = nt * 128;
  const int K = Hh * Dd;

  v8f acc[8];
#pragma unroll
  for (int nb = 0; nb < 8; ++nb) acc[nb] = vzero8f();

  for (int f0 = 0; f0 < K; f0 += 32) {
    const __bf16* ap = a + (long)(row0 + lh) * K + f0 + hi * 8;
    v16bf av = load_frag16(ap, ap + 16);
    v16bf bfr[8];
#pragma unroll
    for (int nb = 0; nb < 8; ++nb) {
      const __bf16* bp = wb + (long)(col0 + nb * 16 + lh) * K + f0 + hi * 8;
      bfr[nb] = load_frag16(bp, bp + 16);
    }
    acc[0] = WMMA_BF16(false, av, false, bfr[0], (short)0, acc[0], false, false);
#pragma unroll
    for (int nb = 1; nb < 8; ++nb)
      acc[nb] = WMMA_BF16(false, av, false, bfr[nb], (short)0, acc[nb], true, false);
  }
#pragma unroll
  for (int nb = 0; nb < 8; ++nb) {
    int col = col0 + nb * 16 + lh;
    float bv = bias[col];
#pragma unroll
    for (int r = 0; r < 8; ++r) {
      int row = row0 + r + hi * 8;
      out[(long)row * OO + col] = acc[nb][r] + bv;
    }
  }
}

// ---------------------------------------------------------------------------
extern "C" void kernel_launch(void* const* d_in, const int* in_sizes, int n_in,
                              void* d_out, int out_size, void* d_ws, size_t ws_size,
                              hipStream_t stream) {
  (void)in_sizes; (void)n_in; (void)out_size; (void)ws_size;
  const float* k   = (const float*)d_in[0];
  const float* q   = (const float*)d_in[1];
  const float* wkx = (const float*)d_in[2];
  const float* wqx = (const float*)d_in[3];
  const float* pw  = (const float*)d_in[4];
  const float* pb  = (const float*)d_in[5];
  // d_in[6] = flag (setup uses False; False path implemented)

  char* ws = (char*)d_ws;
  size_t off = 0;
  auto alloc = [&](size_t bytes) -> char* {
    char* p = ws + off;
    off = (off + bytes + 255) & ~(size_t)255;
    return p;
  };
  __bf16* wkT  = (__bf16*)alloc((size_t)Hh * Dd * Ee * 2);        // 2 MB
  __bf16* wqT  = (__bf16*)alloc((size_t)Hh * Dd * Ee * 2);        // 2 MB
  __bf16* pwb  = (__bf16*)alloc((size_t)OO * Hh * Dd * 2);        // 2 MB
  __bf16* kxb  = (__bf16*)alloc((size_t)Bb * Hh * Ll * Dd * 2);   // 16 MB
  __bf16* qxb  = (__bf16*)alloc((size_t)Bb * Hh * Ll * Dd * 2);   // 16 MB
  __bf16* kxT  = (__bf16*)alloc((size_t)Bb * Hh * Dd * Ll * 2);   // 16 MB
  __bf16* obuf = (__bf16*)alloc((size_t)Bb * Ll * Hh * Dd * 2);   // 16 MB

  // 0) weight conversion / transpose
  {
    long total = 2L * Hh * Ee * Dd + (long)OO * Hh * Dd;
    int blocks = (int)((total + 255) / 256);
    cvt_weights<<<blocks, 256, 0, stream>>>(wkx, wqx, pw, wkT, wqT, pwb);
  }
  // 1) kx = sigmoid(k @ w_kx)  (+ transposed copy),  qx = sigmoid(q @ w_qx)
  {
    int blocks = Bb * Hh * (Ll / 128); // 512
    proj_act<<<blocks, 256, 0, stream>>>(k, wkT, kxb, kxT, 1);
    proj_act<<<blocks, 256, 0, stream>>>(q, wqT, qxb, kxT, 0);
  }
  // 2) flash attention
  {
    int blocks = Bb * Hh * (Ll / 128); // 512
    attn_kernel<<<blocks, 256, 0, stream>>>(qxb, kxb, kxT, obuf);
  }
  // 3) output projection
  {
    int blocks = (Bb * Ll / 128) * (OO / 128); // 512
    proj_kernel<<<blocks, 256, 0, stream>>>(obuf, pwb, pb, (float*)d_out);
  }
}
